// FWHTVectorized_34273839022654
// MI455X (gfx1250) — matrix operand order repaired and checked by reading, not compiled, not run on hardware
//
#include <hip/hip_runtime.h>

typedef __attribute__((ext_vector_type(16))) _Float16 v16h;
typedef __attribute__((ext_vector_type(8)))  float    v8f;
typedef __attribute__((ext_vector_type(4)))  float    v4f;
typedef __attribute__((ext_vector_type(2)))  float    v2f;

#define LDS_STRIDE 68       // 64 + pad: rows are 272B, keeps float4/float2 alignment
#define ROWS_PER_BLOCK 4

// Each length-4096 row viewed as V (64x64, row-major).
// y_row = (1/64) * H64 * V * H64, computed as two stages of S(W) = (W*H64)^T.
// f16 WMMA with exact hi/lo split => f32-accurate, compute stays ~17 GFLOP total,
// far under the 23.3 TB/s HBM roof (~268 MB moved, ~11.5 us).
__global__ __launch_bounds__(128) void fwht4096_wmma_kernel(
    const float* __restrict__ x, float* __restrict__ y) {
  __shared__ float bufA[64 * LDS_STRIDE];
  __shared__ float bufB[64 * LDS_STRIDE];

  const int tid   = threadIdx.x;
  const int lane  = tid & 31;
  const int wave  = tid >> 5;      // 0..3 -> output row-block
  const int grp   = lane >> 4;     // lane group 0/1
  const int lan16 = lane & 15;
  const long row0 = (long)blockIdx.x * ROWS_PER_BLOCK;

  // ---- B fragments: H64 (+/-1) in f16, analytic (sign = parity(popc(K&N))) ----
  // Built ONCE per block, reused for all rows (amortizes ~300 VALU ops 4x).
  // f16 B(32x16) layout: lane holds N = lane%16, K = 2v + h + (v<4?0:8) + 8*(lane/16) (+32*kc)
  v16h Bf[4][2];
  #pragma unroll
  for (int cb = 0; cb < 4; ++cb) {
    #pragma unroll
    for (int kc = 0; kc < 2; ++kc) {
      #pragma unroll
      for (int v = 0; v < 8; ++v) {
        #pragma unroll
        for (int h = 0; h < 2; ++h) {
          int K = 32 * kc + 8 * grp + 2 * v + h + ((v < 4) ? 0 : 8);
          int N = 16 * cb + lan16;
          Bf[cb][kc][2 * v + h] =
              (__popc(K & N) & 1) ? (_Float16)(-1.0f) : (_Float16)(1.0f);
        }
      }
    }
  }

  #pragma unroll 1
  for (int r = 0; r < ROWS_PER_BLOCK; ++r) {
    const float* src = x + (row0 + r) * 4096;
    float*       dst = y + (row0 + r) * 4096;

    // ---- coalesced global -> LDS (row as 64x64, padded stride) ----
    #pragma unroll
    for (int i = 0; i < 8; ++i) {
      int f = (tid + i * 128) * 4;         // flat float index, float4 granularity
      int rr = f >> 6, c = f & 63;
      *(v4f*)(&bufA[rr * LDS_STRIDE + c]) = *(const v4f*)(src + f);
    }

    // Prefetch next row into cache while this one computes (global_prefetch_b8).
    // 128 threads * 128B stride = 16 KB = exactly one row. Uniform branch.
    if (r + 1 < ROWS_PER_BLOCK) {
      __builtin_prefetch(src + 4096 + tid * 32, 0, 1);
    }
    __syncthreads();

    // ---- two identical stages: dst = (src * H)^T ----
    #pragma unroll
    for (int stage = 0; stage < 2; ++stage) {
      const float* sb = (stage == 0) ? bufA : bufB;
      float*       db = (stage == 0) ? bufB : bufA;

      // A fragments for this wave: rows 16*wave..+15, K=0..63, exact hi/lo split.
      // f16 A(16x32) layout: M = lane%16; K = 2v + (v<4?0:8) + 8*(lane/16) (+32*kc)
      v16h a_hi[2], a_lo[2];
      const int M = 16 * wave + lan16;
      #pragma unroll
      for (int kc = 0; kc < 2; ++kc) {
        #pragma unroll
        for (int v = 0; v < 8; ++v) {
          int kbase = 32 * kc + 8 * grp + 2 * v + ((v < 4) ? 0 : 8);
          v2f p = *(const v2f*)(&sb[M * LDS_STRIDE + kbase]);
          _Float16 h0 = (_Float16)p.x;
          _Float16 h1 = (_Float16)p.y;
          a_hi[kc][2 * v]     = h0;
          a_hi[kc][2 * v + 1] = h1;
          a_lo[kc][2 * v]     = (_Float16)(p.x - (float)h0);
          a_lo[kc][2 * v + 1] = (_Float16)(p.y - (float)h1);
        }
      }

      #pragma unroll
      for (int cb = 0; cb < 4; ++cb) {
        v8f acc = {};
        acc = __builtin_amdgcn_wmma_f32_16x16x32_f16(false, a_hi[0], false, Bf[cb][0],
                                                     (short)0, acc, false, false);
        acc = __builtin_amdgcn_wmma_f32_16x16x32_f16(false, a_lo[0], false, Bf[cb][0],
                                                     (short)0, acc, false, false);
        acc = __builtin_amdgcn_wmma_f32_16x16x32_f16(false, a_hi[1], false, Bf[cb][1],
                                                     (short)0, acc, false, false);
        acc = __builtin_amdgcn_wmma_f32_16x16x32_f16(false, a_lo[1], false, Bf[cb][1],
                                                     (short)0, acc, false, false);
        // D(16x16 f32): VGPR g -> M = g + 8*grp, N = lane%16. Store transposed,
        // packing adjacent-M pairs into 8B ds_store_b64 (all offset terms even).
        #pragma unroll
        for (int g = 0; g < 8; g += 2) {
          int m = 16 * wave + g + 8 * grp;
          int n = 16 * cb + lan16;
          v2f p2 = {acc[g], acc[g + 1]};
          *(v2f*)(&db[n * LDS_STRIDE + m]) = p2;
        }
      }
      __syncthreads();
    }

    // ---- coalesced LDS -> global, apply 1/64 normalization ----
    #pragma unroll
    for (int i = 0; i < 8; ++i) {
      int f = (tid + i * 128) * 4;
      int rr = f >> 6, c = f & 63;
      v4f v = *(const v4f*)(&bufA[rr * LDS_STRIDE + c]);
      v *= (1.0f / 64.0f);
      *(v4f*)(dst + f) = v;
    }
    __syncthreads();  // bufA reused as input buffer next iteration
  }
}

extern "C" void kernel_launch(void* const* d_in, const int* in_sizes, int n_in,
                              void* d_out, int out_size, void* d_ws, size_t ws_size,
                              hipStream_t stream) {
  (void)n_in; (void)out_size; (void)d_ws; (void)ws_size;
  const float* x = (const float*)d_in[0];   // (128, 64, 4096) f32
  // d_in[1] = H is deliberately unused: H64 generated analytically in-kernel.
  float* y = (float*)d_out;                 // (128, 64, 4096) f32
  int rows   = in_sizes[0] / 4096;          // 8192
  int blocks = rows / ROWS_PER_BLOCK;       // 2048
  fwht4096_wmma_kernel<<<blocks, 128, 0, stream>>>(x, y);
}
